// CompanyIndustryAttention_80307298501438
// MI455X (gfx1250) — compile-verified, matrix operand order checked by reading, not run on hardware
//
#include <hip/hip_runtime.h>

// ---------------------------------------------------------------------------
// CDNA5 / gfx1250 implementation of CompanyIndustryAttention
// All GEMMs + attention run on v_wmma_f32_16x16x32_bf16 (wave32).
// ---------------------------------------------------------------------------

typedef unsigned short ushort_t;
typedef __attribute__((ext_vector_type(16))) __bf16 v16bf;
typedef __attribute__((ext_vector_type(8)))  float  v8f;

union Frag16 { v16bf v; uint4 q[2]; };

__device__ __forceinline__ ushort_t f2bf(float x) {
  union { float f; unsigned int u; } c; c.f = x;
  unsigned int r = c.u + 0x7FFFu + ((c.u >> 16) & 1u);   // round-to-nearest-even
  return (ushort_t)(r >> 16);
}

// Load a 16x32 bf16 A/B fragment. Caller passes pointer already offset to
// row(lane) * stride + kbase + koff, where koff = (lane>>4)*8.
// Per ISA: VGPR0..3 = elems [koff..koff+7], VGPR4..7 = elems [koff+16..koff+23].
__device__ __forceinline__ v16bf ld_frag(const ushort_t* p) {
  Frag16 f;
  f.q[0] = *(const uint4*)(p);
  f.q[1] = *(const uint4*)(p + 16);
  return f.v;
}

// ---------------------------------------------------------------------------
// fp32 -> bf16 conversion
// ---------------------------------------------------------------------------
__global__ void k_f32_to_bf16(const float* __restrict__ x, ushort_t* __restrict__ y, int n) {
  int i = blockIdx.x * blockDim.x + threadIdx.x;
  if (i < n) y[i] = f2bf(x[i]);
}

// ---------------------------------------------------------------------------
// Zero fill
// ---------------------------------------------------------------------------
__global__ void k_zero_f32(float* __restrict__ p, int n) {
  int i = blockIdx.x * blockDim.x + threadIdx.x;
  if (i < n) p[i] = 0.0f;
}

// ---------------------------------------------------------------------------
// Generic WMMA GEMM:  C[M,N] = A[M (or gather),K] @ B[N,K]^T + bias[N]
// Block = 128 threads (4 waves). Wave computes a 32x64 output tile
// (two 16-row tiles sharing the 4 B fragments). The K loop is unrolled by 2
// with ping-pong fragment buffers: no register rotation, loads for step k+1
// issue while step k's 8 WMMAs execute.  Requires K/32 even (K = 128 or 256).
// grid = (ceil(M/128), N/64).  Optional fp32 and/or bf16 outputs.
// ---------------------------------------------------------------------------
__global__ void __launch_bounds__(128)
k_wmma_gemm_bias(const ushort_t* __restrict__ A,
                 const ushort_t* __restrict__ B,
                 const float*    __restrict__ bias,
                 float*          __restrict__ Cf,
                 ushort_t*       __restrict__ Cb,
                 const long long* __restrict__ gather,
                 int M, int N, int K)
{
  const int lane = threadIdx.x & 31;
  const int wave = threadIdx.x >> 5;
  const int m    = lane & 15;
  const int half = lane >> 4;
  const int koff = half * 8;
  const int row0 = blockIdx.x * 128 + wave * 32;
  const int col0 = blockIdx.y * 64;

  const ushort_t* Ab[2];
#pragma unroll
  for (int t = 0; t < 2; ++t) {
    int ar = row0 + t * 16 + m;
    if (ar > M - 1) ar = M - 1;                  // clamp ragged tail (loads only)
    const long long sr = gather ? gather[ar] : (long long)ar;
    Ab[t] = A + (size_t)sr * K + koff;
  }

  const v8f z = {0.f,0.f,0.f,0.f,0.f,0.f,0.f,0.f};
  v8f acc[2][4];
  const ushort_t* Bp[4];
#pragma unroll
  for (int c = 0; c < 4; ++c) {
    acc[0][c] = z; acc[1][c] = z;
    Bp[c] = B + (size_t)(col0 + c * 16 + m) * K + koff;
  }

#define GEMM_WMMA8(AA0, AA1, BB)                                              \
  _Pragma("unroll")                                                           \
  for (int c = 0; c < 4; ++c) {                                               \
    acc[0][c] = __builtin_amdgcn_wmma_f32_16x16x32_bf16(                      \
        false, AA0, false, BB[c], (short)0, acc[0][c], false, false);         \
    acc[1][c] = __builtin_amdgcn_wmma_f32_16x16x32_bf16(                      \
        false, AA1, false, BB[c], (short)0, acc[1][c], false, false);         \
  }

  // Ping-pong buffers: set X and set Y. No register rotation needed.
  v16bf ax0 = ld_frag(Ab[0]);
  v16bf ax1 = ld_frag(Ab[1]);
  v16bf bx[4];
#pragma unroll
  for (int c = 0; c < 4; ++c) bx[c] = ld_frag(Bp[c]);

  v16bf ay0, ay1, by[4];

  int k = 0;
  for (; k + 64 < K; k += 64) {
    // prefetch step k+32 into Y
    ay0 = ld_frag(Ab[0] + k + 32);
    ay1 = ld_frag(Ab[1] + k + 32);
#pragma unroll
    for (int c = 0; c < 4; ++c) by[c] = ld_frag(Bp[c] + k + 32);

    GEMM_WMMA8(ax0, ax1, bx)                     // compute step k

    // prefetch step k+64 into X
    ax0 = ld_frag(Ab[0] + k + 64);
    ax1 = ld_frag(Ab[1] + k + 64);
#pragma unroll
    for (int c = 0; c < 4; ++c) bx[c] = ld_frag(Bp[c] + k + 64);

    GEMM_WMMA8(ay0, ay1, by)                     // compute step k+32
  }
  // Epilogue: X holds step k; fetch step k+32 into Y, compute both.
  ay0 = ld_frag(Ab[0] + k + 32);
  ay1 = ld_frag(Ab[1] + k + 32);
#pragma unroll
  for (int c = 0; c < 4; ++c) by[c] = ld_frag(Bp[c] + k + 32);
  GEMM_WMMA8(ax0, ax1, bx)
  GEMM_WMMA8(ay0, ay1, by)
#undef GEMM_WMMA8

#pragma unroll
  for (int t = 0; t < 2; ++t) {
#pragma unroll
    for (int c = 0; c < 4; ++c) {
      const int col = col0 + c * 16 + m;          // N index = lane&15
      const float bv = bias ? bias[col] : 0.f;
#pragma unroll
      for (int r = 0; r < 8; ++r) {
        const int rr = row0 + t * 16 + r + 8 * half;   // M index = r + 8*half
        if (rr < M) {
          float v = acc[t][c][r] + bv;
          if (Cf) Cf[(size_t)rr * N + col] = v;
          if (Cb) Cb[(size_t)rr * N + col] = f2bf(v);
        }
      }
    }
  }
}

// ---------------------------------------------------------------------------
// Flash attention, one wave per (16-query block, head).
// Q,K,V: [E, 256] bf16, head h uses columns [h*64, h*64+64).
// ctx out: [E, 256] bf16.  grid = (E/16, H), block = 32.
// ---------------------------------------------------------------------------
__global__ void __launch_bounds__(32)
k_flash_attn(const ushort_t* __restrict__ Q,
             const ushort_t* __restrict__ Kh,
             const ushort_t* __restrict__ V,
             ushort_t* __restrict__ ctx,
             int E, int D, float scale)
{
  __shared__ __align__(16) ushort_t lds_vt[64 * 32];   // V^T block: [d][n]
  __shared__ __align__(16) ushort_t lds_p [16 * 32];   // P block:   [m][n]

  const int lane = threadIdx.x;
  const int m    = lane & 15;
  const int half = lane >> 4;
  const int koff = half * 8;
  const int q0   = blockIdx.x * 16;
  const int hb   = blockIdx.y * 64;

  // Q fragments (16 rows x 64 dims = two 16x32 A frags), kept in registers.
  const ushort_t* qrow = Q + (size_t)(q0 + m) * D + hb;
  const v16bf qf0 = ld_frag(qrow + 0  + koff);
  const v16bf qf1 = ld_frag(qrow + 32 + koff);

  const v8f z = {0.f,0.f,0.f,0.f,0.f,0.f,0.f,0.f};
  v8f o[4] = { z, z, z, z };
  float mrow[8], lrow[8];
#pragma unroll
  for (int r = 0; r < 8; ++r) { mrow[r] = -1e30f; lrow[r] = 0.f; }

  for (int k0 = 0; k0 < E; k0 += 32) {
    // Prefetch next K/V rows (one row per lane) -> global_prefetch_b8.
    if (k0 + 32 < E) {
      __builtin_prefetch(Kh + (size_t)(k0 + 32 + lane) * D + hb, 0, 1);
      __builtin_prefetch(V  + (size_t)(k0 + 32 + lane) * D + hb, 0, 1);
    }

    // Stage V block transposed into LDS: lds_vt[d][n] = V[k0+n][hb+d].
    // 32x64 bf16 = 256 16B chunks; 8 chunks per lane.
#pragma unroll
    for (int p = 0; p < 8; ++p) {
      const int c  = lane + 32 * p;      // chunk id 0..255
      const int n  = c >> 3;             // key within block
      const int d0 = (c & 7) * 8;        // dim chunk base
      uint4 v4 = *(const uint4*)(V + (size_t)(k0 + n) * D + hb + d0);
      const ushort_t* e = (const ushort_t*)&v4;
#pragma unroll
      for (int j = 0; j < 8; ++j)
        lds_vt[(d0 + j) * 32 + n] = e[j];
    }

    // S tiles: S[t][m][n] = sum_d Q[m][d] * K[k0+16t+n][d]
    v8f s[2];
#pragma unroll
    for (int t = 0; t < 2; ++t) {
      const ushort_t* kb = Kh + (size_t)(k0 + t * 16 + m) * D + hb + koff;
      v8f acc = __builtin_amdgcn_wmma_f32_16x16x32_bf16(
          false, qf0, false, ld_frag(kb), (short)0, z, false, false);
      acc = __builtin_amdgcn_wmma_f32_16x16x32_bf16(
          false, qf1, false, ld_frag(kb + 32), (short)0, acc, false, false);
      s[t] = acc;
    }

    // Online softmax. Row of element r is (r + 8*half); row spans lanes with
    // equal `half`, so xor-reductions over masks 1,2,4,8 stay in-row.
    float rmax[8];
#pragma unroll
    for (int r = 0; r < 8; ++r) {
      s[0][r] *= scale; s[1][r] *= scale;
      rmax[r] = fmaxf(s[0][r], s[1][r]);
    }
#pragma unroll
    for (int off = 1; off < 16; off <<= 1)
#pragma unroll
      for (int r = 0; r < 8; ++r)
        rmax[r] = fmaxf(rmax[r], __shfl_xor(rmax[r], off, 32));

    float alpha[8], rsum[8];
#pragma unroll
    for (int r = 0; r < 8; ++r) {
      const float nm = fmaxf(mrow[r], rmax[r]);
      alpha[r] = __expf(mrow[r] - nm);
      mrow[r]  = nm;
      const float p0 = __expf(s[0][r] - nm);
      const float p1 = __expf(s[1][r] - nm);
      s[0][r] = p0; s[1][r] = p1;
      rsum[r] = p0 + p1;
    }
#pragma unroll
    for (int off = 1; off < 16; off <<= 1)
#pragma unroll
      for (int r = 0; r < 8; ++r)
        rsum[r] += __shfl_xor(rsum[r], off, 32);
#pragma unroll
    for (int r = 0; r < 8; ++r) lrow[r] = alpha[r] * lrow[r] + rsum[r];
#pragma unroll
    for (int c = 0; c < 4; ++c)
#pragma unroll
      for (int r = 0; r < 8; ++r) o[c][r] *= alpha[r];

    // P (C/D layout) -> LDS row-major -> re-read as A fragment.
#pragma unroll
    for (int t = 0; t < 2; ++t)
#pragma unroll
      for (int r = 0; r < 8; ++r)
        lds_p[(r + 8 * half) * 32 + t * 16 + m] = f2bf(s[t][r]);

    __syncthreads();   // lds_p + lds_vt visible before fragment reads

    const v16bf pf = ld_frag(lds_p + m * 32 + koff);
#pragma unroll
    for (int c = 0; c < 4; ++c) {
      const v16bf vt = ld_frag(lds_vt + (c * 16 + m) * 32 + koff);
      o[c] = __builtin_amdgcn_wmma_f32_16x16x32_bf16(
          false, pf, false, vt, (short)0, o[c], false, false);
    }

    __syncthreads();   // protect LDS from next-iteration overwrite
  }

  // Epilogue: normalize by row sums, write ctx (bf16).
#pragma unroll
  for (int c = 0; c < 4; ++c) {
    const int col = hb + c * 16 + m;
#pragma unroll
    for (int r = 0; r < 8; ++r) {
      const int rr = q0 + r + 8 * half;
      ctx[(size_t)rr * D + col] = f2bf(o[c][r] / lrow[r]);
    }
  }
}

// ---------------------------------------------------------------------------
// Segment-sum scatter: agg[src[e]] += attn_out[e], counts[src[e]] += 1
// ---------------------------------------------------------------------------
__global__ void k_scatter_mean(const float* __restrict__ attn_out,
                               const long long* __restrict__ src,
                               float* __restrict__ agg,
                               float* __restrict__ counts,
                               int E, int D)
{
  int idx = blockIdx.x * blockDim.x + threadIdx.x;
  if (idx >= E * D) return;
  const int e = idx / D, d = idx - e * D;
  const long long s = src[e];
  atomicAdd(&agg[(size_t)s * D + d], attn_out[idx]);
  if (d == 0) atomicAdd(&counts[s], 1.0f);
}

// ---------------------------------------------------------------------------
// out = LayerNorm(company_h + agg/(counts+1e-6)) * gamma + beta
// One wave per row (D=256 -> 8 elements/lane). Block = 256 (8 waves).
// ---------------------------------------------------------------------------
__global__ void __launch_bounds__(256)
k_finalize_ln(const float* __restrict__ company_h,
              const float* __restrict__ agg,
              const float* __restrict__ counts,
              const float* __restrict__ gamma,
              const float* __restrict__ beta,
              float* __restrict__ out, int M, int D)
{
  const int wave = threadIdx.x >> 5;
  const int lane = threadIdx.x & 31;
  const int row  = blockIdx.x * 8 + wave;
  if (row >= M) return;

  const float inv = 1.0f / (counts[row] + 1e-6f);
  float x[8];
  float s = 0.f;
#pragma unroll
  for (int i = 0; i < 8; ++i) {
    const int d = lane + 32 * i;
    x[i] = company_h[(size_t)row * D + d] + agg[(size_t)row * D + d] * inv;
    s += x[i];
  }
#pragma unroll
  for (int off = 1; off < 32; off <<= 1) s += __shfl_xor(s, off, 32);
  const float mean = s * (1.0f / 256.0f);

  float v = 0.f;
#pragma unroll
  for (int i = 0; i < 8; ++i) { const float dx = x[i] - mean; v += dx * dx; }
#pragma unroll
  for (int off = 1; off < 32; off <<= 1) v += __shfl_xor(v, off, 32);
  const float rstd = rsqrtf(v * (1.0f / 256.0f) + 1e-5f);

#pragma unroll
  for (int i = 0; i < 8; ++i) {
    const int d = lane + 32 * i;
    out[(size_t)row * D + d] = (x[i] - mean) * rstd * gamma[d] + beta[d];
  }
}

// ---------------------------------------------------------------------------
// Host launch
// ---------------------------------------------------------------------------
extern "C" void kernel_launch(void* const* d_in, const int* in_sizes, int n_in,
                              void* d_out, int out_size, void* d_ws, size_t ws_size,
                              hipStream_t stream) {
  (void)in_sizes; (void)n_in; (void)out_size; (void)ws_size;

  constexpr int NC = 20000, NI = 500, E = 8192;
  constexpr int CC = 256, CI = 128, D = 256, H = 4;

  const float*     company_x  = (const float*)d_in[0];
  const float*     industry_x = (const float*)d_in[1];
  const long long* edge_index = (const long long*)d_in[2];
  const float*     Wc    = (const float*)d_in[3];
  const float*     bc    = (const float*)d_in[4];
  const float*     Wi    = (const float*)d_in[5];
  const float*     bi    = (const float*)d_in[6];
  const float*     w_in  = (const float*)d_in[7];
  const float*     b_in  = (const float*)d_in[8];
  const float*     w_out = (const float*)d_in[9];
  const float*     b_out = (const float*)d_in[10];
  const float*     gamma = (const float*)d_in[11];
  const float*     beta  = (const float*)d_in[12];
  float*           out   = (float*)d_out;

  const long long* src = edge_index;        // edge_index[0]
  const long long* tgt = edge_index + E;    // edge_index[1]

  // Workspace bump allocator (256B aligned slices).
  char* w = (char*)d_ws;
  auto alloc = [&](size_t bytes) -> void* {
    void* p = (void*)w;
    w += (bytes + 255) & ~(size_t)255;
    return p;
  };
  ushort_t* cxb   = (ushort_t*)alloc((size_t)NC * CC * 2);   // company_x bf16
  ushort_t* wcb   = (ushort_t*)alloc((size_t)D * CC * 2);    // Wc bf16
  ushort_t* ixb   = (ushort_t*)alloc((size_t)NI * CI * 2);   // industry_x bf16
  ushort_t* wib   = (ushort_t*)alloc((size_t)D * CI * 2);    // Wi bf16
  ushort_t* winb  = (ushort_t*)alloc((size_t)3 * D * D * 2); // w_in bf16
  ushort_t* woutb = (ushort_t*)alloc((size_t)D * D * 2);     // w_out bf16
  float*    ch_f  = (float*)   alloc((size_t)NC * D * 4);    // company_h fp32
  ushort_t* ch_b  = (ushort_t*)alloc((size_t)NC * D * 2);    // company_h bf16
  ushort_t* ih_b  = (ushort_t*)alloc((size_t)NI * D * 2);    // industry_h bf16
  ushort_t* qh    = (ushort_t*)alloc((size_t)E * D * 2);
  ushort_t* kh    = (ushort_t*)alloc((size_t)E * D * 2);
  ushort_t* vh    = (ushort_t*)alloc((size_t)E * D * 2);
  ushort_t* ctxb  = (ushort_t*)alloc((size_t)E * D * 2);
  float*    aout  = (float*)   alloc((size_t)E * D * 4);     // attn_out fp32
  float*    agg   = (float*)   alloc((size_t)NC * D * 4);    // contiguous with counts
  float*    cnts  = (float*)   alloc((size_t)NC * 4);

  auto cvt = [&](const float* s, ushort_t* dptr, int n) {
    k_f32_to_bf16<<<(n + 255) / 256, 256, 0, stream>>>(s, dptr, n);
  };
  cvt(company_x,  cxb,   NC * CC);
  cvt(Wc,         wcb,   D * CC);
  cvt(industry_x, ixb,   NI * CI);
  cvt(Wi,         wib,   D * CI);
  cvt(w_in,       winb,  3 * D * D);
  cvt(w_out,      woutb, D * D);

  // company_h = company_x @ Wc^T + bc   (fp32 + bf16 copies)
  k_wmma_gemm_bias<<<dim3((NC + 127) / 128, D / 64), 128, 0, stream>>>(
      cxb, wcb, bc, ch_f, ch_b, nullptr, NC, D, CC);

  // industry_h = industry_x @ Wi^T + bi (bf16 only)
  k_wmma_gemm_bias<<<dim3((NI + 127) / 128, D / 64), 128, 0, stream>>>(
      ixb, wib, bi, nullptr, ih_b, nullptr, NI, D, CI);

  // Q/K/V projections with fused gathers.
  k_wmma_gemm_bias<<<dim3(E / 128, D / 64), 128, 0, stream>>>(
      ch_b, winb + 0 * D * D, b_in + 0 * D, nullptr, qh, src, E, D, D);
  k_wmma_gemm_bias<<<dim3(E / 128, D / 64), 128, 0, stream>>>(
      ih_b, winb + 1 * D * D, b_in + 1 * D, nullptr, kh, tgt, E, D, D);
  k_wmma_gemm_bias<<<dim3(E / 128, D / 64), 128, 0, stream>>>(
      ih_b, winb + 2 * D * D, b_in + 2 * D, nullptr, vh, tgt, E, D, D);

  // Flash attention: ctx = softmax(Q K^T / sqrt(64)) V, per head.
  k_flash_attn<<<dim3(E / 16, H), 32, 0, stream>>>(
      qh, kh, vh, ctxb, E, D, 0.125f);

  // attn_out = ctx @ w_out^T + b_out (fp32)
  k_wmma_gemm_bias<<<dim3(E / 128, D / 64), 128, 0, stream>>>(
      ctxb, woutb, b_out, aout, nullptr, nullptr, E, D, D);

  // Segment mean: zero, scatter, then fused divide in LN epilogue.
  k_zero_f32<<<(NC * D + NC + 255) / 256, 256, 0, stream>>>(agg, NC * D + NC);
  k_scatter_mean<<<(E * D + 255) / 256, 256, 0, stream>>>(aout, src, agg, cnts, E, D);

  // out = LN(company_h + agg/(counts+1e-6)) * gamma + beta
  k_finalize_ln<<<(NC + 7) / 8, 256, 0, stream>>>(
      ch_f, agg, cnts, gamma, beta, out, NC, D);
}
